// NonLocalBlock_7095285973305
// MI455X (gfx1250) — matrix operand order repaired and checked by reading, not compiled, not run on hardware
//
#include <hip/hip_runtime.h>
#include <hip/hip_bf16.h>

// ---------------------------------------------------------------------------
// NonLocalBlock for MI455X (gfx1250): bf16 WMMA GEMMs + fused LDS attention.
// All GEMM B-matrices are stored n-major (transposed) so every WMMA fragment
// is assembled from two 16-byte vector loads (global_load_b128 class).
// ---------------------------------------------------------------------------

typedef __attribute__((ext_vector_type(16))) __bf16 v16bf;
typedef __attribute__((ext_vector_type(8)))  __bf16 bf16x8;
typedef __attribute__((ext_vector_type(8)))  float  v8f;

#define B_   8
#define C_   512
#define CI_  256
#define N_   6272     // 8*28*28
#define NP_  1568     // 8*14*14
#define NP_TILES 98   // 1568/16
#define EPS_ 1e-5f

union FragU { v16bf v; bf16x8 h[2]; };

// A matrix 16x32 (MxK), row-major source. lane<16: row=lane, K={0..7,16..23};
// lane>=16: row=lane-16, K={8..15,24..31}.   Two b128 loads.
__device__ __forceinline__ v16bf load_a_frag(const __bf16* tile, int ld, int lane) {
    int row = lane & 15;
    int d   = (lane < 16) ? 0 : 8;
    FragU u;
    u.h[0] = *reinterpret_cast<const bf16x8*>(tile + row * ld + d);
    u.h[1] = *reinterpret_cast<const bf16x8*>(tile + row * ld + d + 16);
    return u.v;
}

// B matrix 32x16 (KxN) where the source is n-major: element (k,n) = tile[n*ld+k].
// lane<16: col=lane, K={0..7,16..23}; lane>=16: col=lane-16, K={8..15,24..31}.  Two b128 loads.
__device__ __forceinline__ v16bf load_b_frag_nmajor(const __bf16* tile, int ld, int lane) {
    int col = lane & 15;
    int kb  = (lane < 16) ? 0 : 8;
    FragU u;
    u.h[0] = *reinterpret_cast<const bf16x8*>(tile + col * ld + kb);
    u.h[1] = *reinterpret_cast<const bf16x8*>(tile + col * ld + kb + 16);
    return u.v;
}

// ---------------- transpose + convert --------------------------------------
// x [B][C][N] f32  ->  xT [B][N][C] bf16   (LDS tiled, coalesced both sides)
__launch_bounds__(256)
__global__ void transpose_bf16_kernel(const float* __restrict__ x,
                                      __bf16* __restrict__ xT) {
    __shared__ float tile[32][33];
    int b  = blockIdx.z;
    int n0 = blockIdx.x * 32, c0 = blockIdx.y * 32;
    const float* xb  = x  + (long long)b * C_ * N_;
    __bf16*      xtb = xT + (long long)b * N_ * C_;
#pragma unroll
    for (int i = threadIdx.y; i < 32; i += 8)
        tile[i][threadIdx.x] = xb[(long long)(c0 + i) * N_ + n0 + threadIdx.x];
    __syncthreads();
#pragma unroll
    for (int i = threadIdx.y; i < 32; i += 8)
        xtb[(long long)(n0 + i) * C_ + c0 + threadIdx.x] = (__bf16)tile[threadIdx.x][i];
}

// 2x2 maxpool -> c-major bf16 output  mpT [B][NP][C]  (coalesced stores)
__global__ void maxpool_t_kernel(const float* __restrict__ x, __bf16* __restrict__ mpT) {
    long long idx = (long long)blockIdx.x * blockDim.x + threadIdx.x;
    const long long OUT = (long long)B_ * NP_ * C_;
    if (idx >= OUT) return;
    int c  = (int)(idx % C_);
    long long tmp = idx / C_;
    int np = (int)(tmp % NP_);
    int b  = (int)(tmp / NP_);
    int tt = np / 196, r = np % 196, i = r / 14, j = r % 14;
    const float* s = x + (((long long)b * C_ + c) * 8 + tt) * 784 + i * 56 + j * 2;
    float m = fmaxf(fmaxf(s[0], s[1]), fmaxf(s[28], s[29]));
    mpT[idx] = (__bf16)m;
}

__global__ void f32_to_bf16_kernel(const float* __restrict__ in,
                                   __bf16* __restrict__ out, long long n) {
    long long i = (long long)blockIdx.x * blockDim.x + threadIdx.x;
    if (i < n) out[i] = (__bf16)in[i];
}

// ---------------- generic WMMA projection GEMM -----------------------------
// Out[b] = W[MxK] * X[b]  where X is given n-major: XT [N][K].
// One wave computes a 16(M) x 32(N) tile (A fragment reused twice).
// mode 0: bf16 store [M][N]  mode 1: bf16 transposed store [N][M] (vector)
// mode 2: f32 store [M][N]
__launch_bounds__(32)
__global__ void proj_wmma_kernel(const __bf16* __restrict__ W,
                                 const __bf16* __restrict__ XT,
                                 const float*  __restrict__ bias,
                                 void* __restrict__ out,
                                 int M, int K, int N,
                                 long long strideX, long long strideOut, int mode) {
    int lane = threadIdx.x;
    int nt = blockIdx.x * 2;   // two 16-wide n-tiles
    int mt = blockIdx.y, b = blockIdx.z;
    const __bf16* Xb = XT + (long long)b * strideX;

    v8f acc0 = {}, acc1 = {};
    for (int kk = 0; kk < K; kk += 32) {
        v16bf a  = load_a_frag(W + mt * 16 * K + kk, K, lane);
        v16bf b0 = load_b_frag_nmajor(Xb + (long long)(nt * 16) * K + kk, K, lane);
        v16bf b1 = load_b_frag_nmajor(Xb + (long long)(nt * 16 + 16) * K + kk, K, lane);
        acc0 = __builtin_amdgcn_wmma_f32_16x16x32_bf16(false, a, false, b0,
                                                       (short)0, acc0, false, false);
        acc1 = __builtin_amdgcn_wmma_f32_16x16x32_bf16(false, a, false, b1,
                                                       (short)0, acc1, false, false);
    }
    int col = lane & 15;
    int rb  = (lane < 16) ? 0 : 8;
    int m0  = mt * 16;
#pragma unroll
    for (int half = 0; half < 2; ++half) {
        v8f acc = half ? acc1 : acc0;
        int n = (nt + half) * 16 + col;
        if (mode == 1) {
            bf16x8 vst;
#pragma unroll
            for (int r = 0; r < 8; ++r) vst[r] = (__bf16)(acc[r] + bias[m0 + rb + r]);
            *reinterpret_cast<bf16x8*>(
                (__bf16*)out + (long long)b * strideOut + (long long)n * M + m0 + rb) = vst;
        } else if (mode == 0) {
#pragma unroll
            for (int r = 0; r < 8; ++r) {
                int m = m0 + rb + r;
                ((__bf16*)out)[(long long)b * strideOut + (long long)m * N + n] =
                    (__bf16)(acc[r] + bias[m]);
            }
        } else {
#pragma unroll
            for (int r = 0; r < 8; ++r) {
                int m = m0 + rb + r;
                ((float*)out)[(long long)b * strideOut + (long long)m * N + n] =
                    acc[r] + bias[m];
            }
        }
    }
}

// ---------------- fused attention: scores -> softmax -> t ------------------
// Block = 256 threads (8 waves): one batch b, one 16-row strip of N.
// Scores (16 x 1568) live only in LDS.  theta A-strip is register-resident.
__launch_bounds__(256)
__global__ void attn_kernel(const __bf16* __restrict__ thetaT,  // [B][N][CI]
                            const __bf16* __restrict__ phiT,    // [B][NP][CI]
                            const __bf16* __restrict__ g,       // [B][CI][NP]
                            __bf16* __restrict__ tT) {           // [B][N][CI]
    __shared__ __bf16 sP[16 * NP_];  // 50,176 B

    int b   = blockIdx.y;
    int n0  = blockIdx.x * 16;
    int tid = threadIdx.x;
    int wave = tid >> 5;
    int lane = tid & 31;

    const __bf16* th = thetaT + ((long long)b * N_ + n0) * CI_;
    const __bf16* ph = phiT + (long long)b * NP_ * CI_;
    const __bf16* gg = g    + (long long)b * CI_ * NP_;

    // Preload the 16 x 256 theta A-strip into registers (8 fragments).
    v16bf afr[8];
#pragma unroll
    for (int k8 = 0; k8 < 8; ++k8) afr[k8] = load_a_frag(th + k8 * 32, CI_, lane);

    // ---- Phase 1: scores[16 x NP] = theta_strip * phi^T, scaled by 1/16
    for (int j = wave; j < NP_TILES; j += 8) {
        v8f acc = {};
#pragma unroll
        for (int k8 = 0; k8 < 8; ++k8) {
            v16bf bb = load_b_frag_nmajor(ph + (long long)(j * 16) * CI_ + k8 * 32, CI_, lane);
            acc = __builtin_amdgcn_wmma_f32_16x16x32_bf16(false, afr[k8], false, bb,
                                                          (short)0, acc, false, false);
        }
        int col = lane & 15;
        int rb  = (lane < 16) ? 0 : 8;
#pragma unroll
        for (int r = 0; r < 8; ++r)
            sP[(rb + r) * NP_ + j * 16 + col] = (__bf16)(acc[r] * 0.0625f); // CI^-0.5
    }
    __syncthreads();

    // ---- Phase 2: row softmax over NP (16 threads/row, shfl width 16)
    {
        int row = tid >> 4;
        int l16 = tid & 15;
        __bf16* rp = sP + row * NP_;
        float mx = -3.0e38f;
        for (int i = l16; i < NP_; i += 16) mx = fmaxf(mx, (float)rp[i]);
#pragma unroll
        for (int m = 8; m >= 1; m >>= 1) mx = fmaxf(mx, __shfl_xor(mx, m, 16));
        float sum = 0.f;
        for (int i = l16; i < NP_; i += 16) {
            float e = __expf((float)rp[i] - mx);
            sum += e;
            rp[i] = (__bf16)e;
        }
#pragma unroll
        for (int m = 8; m >= 1; m >>= 1) sum += __shfl_xor(sum, m, 16);
        float inv = 1.0f / sum;
        for (int i = l16; i < NP_; i += 16) rp[i] = (__bf16)((float)rp[i] * inv);
    }
    __syncthreads();

    // ---- Phase 3: t_strip = G (CIxNP) * P^T (NPx16); wave w -> ci tiles 2w,2w+1
    for (int ct = wave * 2; ct < wave * 2 + 2; ++ct) {
        v8f acc = {};
        for (int m2 = 0; m2 < NP_ / 32; ++m2) {
            v16bf a  = load_a_frag(gg + ct * 16 * NP_ + m2 * 32, NP_, lane);
            v16bf bb = load_b_frag_nmajor(sP + m2 * 32, NP_, lane);  // P^T from LDS
            acc = __builtin_amdgcn_wmma_f32_16x16x32_bf16(false, a, false, bb,
                                                          (short)0, acc, false, false);
        }
        int col = lane & 15;
        int rb  = (lane < 16) ? 0 : 8;
        bf16x8 vst;
#pragma unroll
        for (int r = 0; r < 8; ++r) vst[r] = (__bf16)acc[r];
        // tT[n0+col][ct*16+rb .. +7] : one 16B store per lane
        *reinterpret_cast<bf16x8*>(
            tT + ((long long)b * N_ + n0 + col) * CI_ + ct * 16 + rb) = vst;
    }
}

// ---------------- BatchNorm (training-mode batch stats) --------------------
__launch_bounds__(256)
__global__ void bn_stats_kernel(const float* __restrict__ o,
                                float* __restrict__ mean, float* __restrict__ invstd) {
    int c = blockIdx.x, tid = threadIdx.x;
    __shared__ float s1[256], s2[256];
    float sum = 0.f, sq = 0.f;
    for (int b = 0; b < B_; ++b) {
        const float* p = o + ((long long)b * C_ + c) * N_;
        for (int n = tid; n < N_; n += 256) { float v = p[n]; sum += v; sq += v * v; }
    }
    s1[tid] = sum; s2[tid] = sq;
    __syncthreads();
    for (int s = 128; s > 0; s >>= 1) {
        if (tid < s) { s1[tid] += s1[tid + s]; s2[tid] += s2[tid + s]; }
        __syncthreads();
    }
    if (tid == 0) {
        const float invCnt = 1.0f / ((float)B_ * (float)N_);
        float m   = s1[0] * invCnt;
        float var = s2[0] * invCnt - m * m;
        mean[c]   = m;
        invstd[c] = rsqrtf(var + EPS_);
    }
}

__global__ void bn_final_kernel(const float* __restrict__ o, const float* __restrict__ x,
                                const float* __restrict__ mean, const float* __restrict__ invstd,
                                const float* __restrict__ gamma, const float* __restrict__ beta,
                                float* __restrict__ out) {
    long long i = (long long)blockIdx.x * blockDim.x + threadIdx.x;
    const long long TOT = (long long)B_ * C_ * N_;
    if (i >= TOT) return;
    int c = (int)((i / N_) % C_);
    out[i] = (o[i] - mean[c]) * invstd[c] * gamma[c] + beta[c] + x[i];
}

// ---------------------------------------------------------------------------
extern "C" void kernel_launch(void* const* d_in, const int* in_sizes, int n_in,
                              void* d_out, int out_size, void* d_ws, size_t ws_size,
                              hipStream_t stream) {
    (void)in_sizes; (void)n_in; (void)out_size; (void)ws_size;
    const float* x       = (const float*)d_in[0];
    const float* theta_w = (const float*)d_in[1];
    const float* theta_b = (const float*)d_in[2];
    const float* phi_w   = (const float*)d_in[3];
    const float* phi_b   = (const float*)d_in[4];
    const float* g_w     = (const float*)d_in[5];
    const float* g_b     = (const float*)d_in[6];
    const float* out_w   = (const float*)d_in[7];
    const float* out_b   = (const float*)d_in[8];
    const float* gamma   = (const float*)d_in[9];
    const float* beta    = (const float*)d_in[10];
    float* out = (float*)d_out;

    char* p = (char*)d_ws;
    auto carve = [&](size_t bytes) -> void* {
        void* r = (void*)p;
        p += (bytes + 255) & ~(size_t)255;
        return r;
    };
    __bf16* xT     = (__bf16*)carve((size_t)B_ * N_ * C_ * 2);     // [B][N][C]
    __bf16* mpT    = (__bf16*)carve((size_t)B_ * NP_ * C_ * 2);    // [B][NP][C]
    __bf16* wth    = (__bf16*)carve((size_t)CI_ * C_ * 2);
    __bf16* wph    = (__bf16*)carve((size_t)CI_ * C_ * 2);
    __bf16* wg     = (__bf16*)carve((size_t)CI_ * C_ * 2);
    __bf16* wout   = (__bf16*)carve((size_t)C_ * CI_ * 2);
    __bf16* thetaT = (__bf16*)carve((size_t)B_ * N_ * CI_ * 2);    // [B][N][CI]
    __bf16* phiT   = (__bf16*)carve((size_t)B_ * NP_ * CI_ * 2);   // [B][NP][CI]
    __bf16* gb     = (__bf16*)carve((size_t)B_ * CI_ * NP_ * 2);   // [B][CI][NP]
    __bf16* tTb    = (__bf16*)carve((size_t)B_ * N_ * CI_ * 2);    // [B][N][CI]
    float*  obuf   = (float*)carve((size_t)B_ * C_ * N_ * 4);      // [B][C][N]
    float*  meanv  = (float*)carve((size_t)C_ * 4);
    float*  istdv  = (float*)carve((size_t)C_ * 4);

    const long long totX  = (long long)B_ * C_ * N_;
    const long long totW  = (long long)CI_ * C_;
    const long long totMP = (long long)B_ * NP_ * C_;

    // 1) transpose/convert + maxpool + weight conversion
    transpose_bf16_kernel<<<dim3(N_ / 32, C_ / 32, B_), dim3(32, 8), 0, stream>>>(x, xT);
    maxpool_t_kernel<<<(unsigned)((totMP + 255) / 256), 256, 0, stream>>>(x, mpT);
    f32_to_bf16_kernel<<<(unsigned)((totW + 255) / 256), 256, 0, stream>>>(theta_w, wth, totW);
    f32_to_bf16_kernel<<<(unsigned)((totW + 255) / 256), 256, 0, stream>>>(phi_w, wph, totW);
    f32_to_bf16_kernel<<<(unsigned)((totW + 255) / 256), 256, 0, stream>>>(g_w, wg, totW);
    f32_to_bf16_kernel<<<(unsigned)((totW + 255) / 256), 256, 0, stream>>>(out_w, wout, totW);

    // 2) projections (WMMA).  theta -> [N][CI] (mode 1), phi -> [NP][CI] (mode 1),
    //    g -> [CI][NP] (mode 0)
    proj_wmma_kernel<<<dim3(N_ / 32, CI_ / 16, B_), 32, 0, stream>>>(
        wth, xT, theta_b, thetaT, CI_, C_, N_,
        (long long)N_ * C_, (long long)N_ * CI_, 1);
    proj_wmma_kernel<<<dim3(NP_ / 32, CI_ / 16, B_), 32, 0, stream>>>(
        wph, mpT, phi_b, phiT, CI_, C_, NP_,
        (long long)NP_ * C_, (long long)NP_ * CI_, 1);
    proj_wmma_kernel<<<dim3(NP_ / 32, CI_ / 16, B_), 32, 0, stream>>>(
        wg, mpT, g_b, gb, CI_, C_, NP_,
        (long long)NP_ * C_, (long long)CI_ * NP_, 0);

    // 3) fused attention (scores + softmax + t), t stored transposed [N][CI]
    attn_kernel<<<dim3(N_ / 16, B_), 256, 0, stream>>>(thetaT, phiT, gb, tTb);

    // 4) output projection: [C x CI] * t -> f32 obuf [C][N] (mode 2)
    proj_wmma_kernel<<<dim3(N_ / 32, C_ / 16, B_), 32, 0, stream>>>(
        wout, tTb, out_b, obuf, C_, CI_, N_,
        (long long)N_ * CI_, (long long)C_ * N_, 2);

    // 5) BatchNorm (batch stats) + residual
    bn_stats_kernel<<<C_, 256, 0, stream>>>(obuf, meanv, istdv);
    bn_final_kernel<<<(unsigned)((totX + 255) / 256), 256, 0, stream>>>(
        obuf, x, meanv, istdv, gamma, beta, out);
}
